// Experts_57466662420619
// MI455X (gfx1250) — matrix operand order repaired and checked by reading, not compiled, not run on hardware
//
#include <hip/hip_runtime.h>
#include <hip/hip_bf16.h>
#include <math.h>

// MoE FFN: out = (gelu_exact(X @ W1[e] + b1[e])) @ W2[e] + b2[e], per-expert row blocks.
// Split-precision bf16 WMMA (hi+lo planes, 3 products) ~= fp32 accuracy, fp32 accumulate.
// Intermediate H (16384x4096 f32 = 256 MiB) in d_ws. Double-buffered LDS, 1 barrier/K-step.

typedef __attribute__((ext_vector_type(16))) __bf16       v16bf;
typedef __attribute__((ext_vector_type(2)))  __bf16       v2bf;
typedef __attribute__((ext_vector_type(8)))  float        v8f;
typedef __attribute__((ext_vector_type(4)))  unsigned int v4u;

#define NE     8
#define DMODEL 1024
#define DFF    4096
#define TPE    2048   // tokens per expert

#define BM 128
#define BN 128
#define BK 32
#define NTHREADS 256

// LDS: per buffer, 4 planes of [128 rows][16 k-pairs] u32, row stride 20
// (pad keeps 16B alignment for b128 and is bank-conflict-free: gcd(20,64)=4).
#define RS    20
#define PLANE (128 * RS)        // 2560 u32
#define AHI   0
#define ALO   (1 * PLANE)
#define BHI   (2 * PLANE)
#define BLO   (3 * PLANE)
#define BUFU  (4 * PLANE)       // 10240 u32 = 40 KB per buffer; x2 = 80 KB

union Frag { v16bf bf; v4u q[2]; };

// Pack two f32 -> one u32 of two bf16 (clang matches this to v_cvt_pk_bf16_f32).
__device__ __forceinline__ unsigned bfpack(float x0, float x1) {
  v2bf p = {(__bf16)x0, (__bf16)x1};
  return __builtin_bit_cast(unsigned, p);
}

// Split two f32 into bf16-hi pair (return) and bf16-lo residual pair (out).
__device__ __forceinline__ unsigned pack2(float x0, float x1, unsigned &lo) {
  unsigned hi = bfpack(x0, x1);
  float h0 = __builtin_bit_cast(float, hi << 16);
  float h1 = __builtin_bit_cast(float, hi & 0xffff0000u);
  lo = bfpack(x0 - h0, x1 - h1);
  return hi;
}

__device__ __forceinline__ void prefetch_tiles(const float* __restrict__ Ae,
                                               const float* __restrict__ Be,
                                               int K, int N, int n0, int tid, int k0,
                                               float2* aReg, float* bReg0, float* bReg1) {
#pragma unroll
  for (int j = 0; j < 8; ++j) {
    int pid = tid + j * NTHREADS;
    // A tile: [128 rows][32 K], one float2 along K per slot
    int ar = pid >> 4, at = pid & 15;
    aReg[j] = *(const float2*)(Ae + (size_t)ar * K + (size_t)(k0 + 2 * at));
    // B tile: [32 K][128 N]; fetch (k,n) and (k+1,n) to form a K-pair
    int bn = pid & 127, bt = pid >> 7;
    const float* bp = Be + (size_t)(k0 + 2 * bt) * N + (size_t)(n0 + bn);
    bReg0[j] = bp[0];
    bReg1[j] = bp[N];
  }
}

template <int GELU>
__global__ __launch_bounds__(NTHREADS)
void moe_gemm(const float* __restrict__ A, const float* __restrict__ B,
              const float* __restrict__ bias, float* __restrict__ C,
              int Mpe, int K, int N) {
  __shared__ unsigned smem[2 * BUFU];   // 80 KB, double-buffered

  const int e  = blockIdx.z;
  const float* Ae = A + ((size_t)e * Mpe + (size_t)blockIdx.y * BM) * (size_t)K;
  const float* Be = B + (size_t)e * K * N;
  const float* be = bias + (size_t)e * N;
  float*       Ce = C + ((size_t)e * Mpe + (size_t)blockIdx.y * BM) * (size_t)N;
  const int n0 = blockIdx.x * BN;

  const int tid   = threadIdx.x;
  const int lane  = tid & 31;
  const int wid   = tid >> 5;   // 8 waves
  const int waveM = wid >> 2;   // 0..1 -> 64 rows each
  const int waveN = wid & 3;    // 0..3 -> 32 cols each
  const int half  = lane >> 4;  // lane half selects K-run per ISA layout
  const int rlo   = lane & 15;

  v8f acc[4][2];
  {
    v8f z = {0.f, 0.f, 0.f, 0.f, 0.f, 0.f, 0.f, 0.f};
#pragma unroll
    for (int mi = 0; mi < 4; ++mi)
#pragma unroll
      for (int ni = 0; ni < 2; ++ni) acc[mi][ni] = z;
  }

  // staging registers for the prefetched global tile
  float2 aReg[8];
  float  bReg0[8], bReg1[8];

  // prefetch + stage K-step 0 into buffer 0
  prefetch_tiles(Ae, Be, K, N, n0, tid, 0, aReg, bReg0, bReg1);
#pragma unroll
  for (int j = 0; j < 8; ++j) {
    int pid = tid + j * NTHREADS;
    int ar = pid >> 4, at = pid & 15;
    unsigned alo, ahi = pack2(aReg[j].x, aReg[j].y, alo);
    smem[AHI + ar * RS + at] = ahi;
    smem[ALO + ar * RS + at] = alo;
    int bn = pid & 127, bt = pid >> 7;
    unsigned blo, bhi = pack2(bReg0[j], bReg1[j], blo);
    smem[BHI + bn * RS + bt] = bhi;   // B stored transposed: [n][k-pair]
    smem[BLO + bn * RS + bt] = blo;
  }
  __syncthreads();

  const int nK = K / BK;
  for (int ks = 0; ks < nK; ++ks) {
    const int cur = (ks & 1) * BUFU;
    const int nxt = ((ks + 1) & 1) * BUFU;
    const bool more = (ks + 1 < nK);

    // issue next tile's global loads first (latency hidden under WMMAs)
    if (more)
      prefetch_tiles(Ae, Be, K, N, n0, tid, (ks + 1) * BK, aReg, bReg0, bReg1);

    // ---- fragment loads from current buffer (ISA 16-bit wave32 layouts) ----
    Frag aH[4], aL[4], bH[2], bL[2];
#pragma unroll
    for (int mi = 0; mi < 4; ++mi) {
      int row = waveM * 64 + mi * 16 + rlo;
      const v4u* ph = (const v4u*)&smem[cur + AHI + row * RS + half * 4];
      aH[mi].q[0] = ph[0]; aH[mi].q[1] = ph[2];     // K 0-7 & 16-23 (or +8)
      const v4u* pl = (const v4u*)&smem[cur + ALO + row * RS + half * 4];
      aL[mi].q[0] = pl[0]; aL[mi].q[1] = pl[2];
    }
#pragma unroll
    for (int ni = 0; ni < 2; ++ni) {
      int col = waveN * 32 + ni * 16 + rlo;
      const v4u* ph = (const v4u*)&smem[cur + BHI + col * RS + half * 4];
      bH[ni].q[0] = ph[0]; bH[ni].q[1] = ph[2];     // K 0-7 & 16-23 (or +8)
      const v4u* pl = (const v4u*)&smem[cur + BLO + col * RS + half * 4];
      bL[ni].q[0] = pl[0]; bL[ni].q[1] = pl[2];
    }

    // ---- 3-product split-precision accumulation: 24 WMMAs / wave / K-step ----
#pragma unroll
    for (int mi = 0; mi < 4; ++mi)
#pragma unroll
      for (int ni = 0; ni < 2; ++ni) {
        v8f c = acc[mi][ni];
        c = __builtin_amdgcn_wmma_f32_16x16x32_bf16(false, aH[mi].bf, false, bH[ni].bf,
                                                    (short)0, c, false, false);
        c = __builtin_amdgcn_wmma_f32_16x16x32_bf16(false, aH[mi].bf, false, bL[ni].bf,
                                                    (short)0, c, false, false);
        c = __builtin_amdgcn_wmma_f32_16x16x32_bf16(false, aL[mi].bf, false, bH[ni].bf,
                                                    (short)0, c, false, false);
        acc[mi][ni] = c;
      }

    // ---- convert + store next tile into the other buffer (overlaps WMMAs) ----
    if (more) {
#pragma unroll
      for (int j = 0; j < 8; ++j) {
        int pid = tid + j * NTHREADS;
        int ar = pid >> 4, at = pid & 15;
        unsigned alo, ahi = pack2(aReg[j].x, aReg[j].y, alo);
        smem[nxt + AHI + ar * RS + at] = ahi;
        smem[nxt + ALO + ar * RS + at] = alo;
        int bn = pid & 127, bt = pid >> 7;
        unsigned blo, bhi = pack2(bReg0[j], bReg1[j], blo);
        smem[nxt + BHI + bn * RS + bt] = bhi;
        smem[nxt + BLO + bn * RS + bt] = blo;
      }
    }
    __syncthreads();   // single barrier per K-step
  }

  // ---- epilogue: bias (+ exact GELU) -> nontemporal f32 store (write-once stream) ----
#pragma unroll
  for (int ni = 0; ni < 2; ++ni) {
    int col = n0 + waveN * 32 + ni * 16 + rlo;
    float bv = be[col];
#pragma unroll
    for (int mi = 0; mi < 4; ++mi) {
      int lrow = waveM * 64 + mi * 16 + half * 8;  // C layout: VGPR r -> M = r (+8 hi lanes)
#pragma unroll
      for (int r = 0; r < 8; ++r) {
        float y = acc[mi][ni][r] + bv;
        if (GELU) y = 0.5f * y * (1.0f + erff(y * 0.70710678118654752f));
        __builtin_nontemporal_store(y, &Ce[(size_t)(lrow + r) * N + col]);
      }
    }
  }
}

extern "C" void kernel_launch(void* const* d_in, const int* in_sizes, int n_in,
                              void* d_out, int out_size, void* d_ws, size_t ws_size,
                              hipStream_t stream) {
  (void)in_sizes; (void)n_in; (void)out_size; (void)ws_size;
  // setup_inputs order: inputs, splits(int64, unused: fixed 2048/expert), W1, b1, W2, b2
  const float* inputs = (const float*)d_in[0];
  const float* W1     = (const float*)d_in[2];
  const float* b1     = (const float*)d_in[3];
  const float* W2     = (const float*)d_in[4];
  const float* b2     = (const float*)d_in[5];
  float* H   = (float*)d_ws;   // needs 16384*4096*4 = 256 MiB scratch
  float* out = (float*)d_out;

  dim3 blk(NTHREADS);
  dim3 g1(DFF / BN, TPE / BM, NE);     // (32, 16, 8)
  hipLaunchKernelGGL((moe_gemm<1>), g1, blk, 0, stream,
                     inputs, W1, b1, H, TPE, DMODEL, DFF);
  dim3 g2(DMODEL / BN, TPE / BM, NE);  // (8, 16, 8)
  hipLaunchKernelGGL((moe_gemm<0>), g2, blk, 0, stream,
                     H, W2, b2, out, TPE, DFF, DMODEL);
}
